// SparseNet_6193342841432
// MI455X (gfx1250) — matrix operand order repaired and checked by reading, not compiled, not run on hardware
//
#include <hip/hip_runtime.h>

// ---------------------------------------------------------------------------
// ISTA sparse coding on gfx1250: bf16 WMMA (hi/lo split-3 ~= f32 accuracy) with
// double-buffered GLOBAL_LOAD_ASYNC_TO_LDS_B128 staging (ASYNCcnt pipelined).
//   B=4096 rows, K=256 codes, P=1024 pixels.
//   iterate: res = img - R@U^T ; G = res@U ; R = soft(R + 0.2*G, 0.005)
//   stop when ||dR||/||R|| < 0.01 (device flag, fixed 256-launch cap)
//   output: R@U^T  (f32)
// ---------------------------------------------------------------------------

typedef __attribute__((ext_vector_type(16))) __bf16         v16bf;
typedef __attribute__((ext_vector_type(8)))  float          v8f;
typedef __attribute__((ext_vector_type(8)))  unsigned short us8;

#define NB    4096
#define NK    256
#define NP    1024
#define LR2   0.2f      // 2 * R_LR
#define LMDA  0.005f
#define TOL2  1.0e-4f   // TOL^2
#define MAXIT 256

static __device__ __forceinline__ unsigned short f2bf(float f) {
    unsigned u = __float_as_uint(f);
    unsigned r = u + 0x7FFFu + ((u >> 16) & 1u);   // RNE to bf16
    return (unsigned short)(r >> 16);
}
static __device__ __forceinline__ float bf2f(unsigned short h) {
    return __uint_as_float(((unsigned)h) << 16);
}

static __device__ __forceinline__ v8f wmma_bf16(v16bf a, v16bf b, v8f c) {
    return __builtin_amdgcn_wmma_f32_16x16x32_bf16(false, a, false, b,
                                                   (short)0, c, false, false);
}

// A fragment 16x32 bf16 from LDS tile stored [row][32] (K contiguous).
// lane l: row = rowBase + (l&15); halves 0..7 = K kb..kb+7, 8..15 = kb+16..kb+23,
// kb = (l>=16)?8:0  (per ISA 16-bit A layout table).
static __device__ __forceinline__ v16bf load_a_frag(const unsigned short* lds,
                                                    int lane, int rowBase) {
    int r  = rowBase + (lane & 15);
    int kb = (lane >> 4) * 8;
    union { v16bf v; us8 h[2]; } f;
    f.h[0] = *(const us8*)(lds + r * 32 + kb);
    f.h[1] = *(const us8*)(lds + r * 32 + kb + 16);
    return f.v;
}

// B fragment 32x16 bf16 from LDS tile stored [col][32] (K contiguous per col).
// lane l: col = colBase + (l&15); halves 0..7 = K kb..kb+7, 8..15 = kb+16..kb+23, kb = (l>=16)?8:0.
static __device__ __forceinline__ v16bf load_b_frag(const unsigned short* lds,
                                                    int lane, int colBase) {
    int n  = colBase + (lane & 15);
    int kb = (lane >> 4) * 8;
    union { v16bf v; us8 h[2]; } f;
    const us8* p = (const us8*)(lds + n * 32 + kb);
    f.h[0] = p[0];
    f.h[1] = p[2];
    return f.v;
}

// Async 16B copy global -> LDS, tracked by ASYNCcnt (bypasses VGPRs).
// Generic shared pointers carry the LDS byte offset in addr[31:0] (flat-LDS
// aperture), which is exactly what VDST of the async op expects.
static __device__ __forceinline__ void async_cp16(unsigned lds_off, const void* g) {
    asm volatile("global_load_async_to_lds_b128 %0, %1, off"
                 :: "v"(lds_off), "v"(g) : "memory");
}

// ---------------------------------------------------------------------------
// 64x64 C-tile GEMM core, K in steps of 32, split-3 bf16 (hi*hi+hi*lo+lo*hi),
// double-buffered async LDS staging. 256 threads = 8 waves (4 x 2 wave grid),
// each wave owns two 16x16 f32 accumulators.
// ---------------------------------------------------------------------------
static __device__ __forceinline__ void gemm64x64(
    const unsigned short* __restrict__ Ah, const unsigned short* __restrict__ Al,
    int strideA, int aRow0,
    const unsigned short* __restrict__ Bh, const unsigned short* __restrict__ Bl,
    int strideB, int bRow0,
    int Ktot, v8f acc[2]) {
    __shared__ unsigned short sAh[2][2048], sAl[2][2048];
    __shared__ unsigned short sBh[2][2048], sBl[2][2048];

    const int tid  = threadIdx.x;
    const int lane = tid & 31;
    const int w    = tid >> 5;
    const int wm   = w & 3;        // 0..3  (M wave tile)
    const int wn   = w >> 2;       // 0..1  (N wave tile, 32 cols each)
    const int r    = tid >> 2;     // staging row 0..63
    const int c    = tid & 3;      // staging 16B chunk 0..3

    const unsigned short* gAh = Ah + (size_t)(aRow0 + r) * strideA + c * 8;
    const unsigned short* gAl = Al + (size_t)(aRow0 + r) * strideA + c * 8;
    const unsigned short* gBh = Bh + (size_t)(bRow0 + r) * strideB + c * 8;
    const unsigned short* gBl = Bl + (size_t)(bRow0 + r) * strideB + c * 8;

    auto issue = [&](int buf, int k0) {
        unsigned d = (unsigned)(tid * 8);
        async_cp16((unsigned)(uintptr_t)&sAh[buf][d], gAh + k0);
        async_cp16((unsigned)(uintptr_t)&sAl[buf][d], gAl + k0);
        async_cp16((unsigned)(uintptr_t)&sBh[buf][d], gBh + k0);
        async_cp16((unsigned)(uintptr_t)&sBl[buf][d], gBl + k0);
    };

    const int steps = Ktot / 32;
    issue(0, 0);
    for (int k = 0; k < steps; ++k) {
        const int buf = k & 1;
        if (k + 1 < steps) {
            issue(buf ^ 1, (k + 1) * 32);             // prefetch next K tile
            // async loads retire in order: <=4 outstanding => oldest group done
            asm volatile("s_wait_asynccnt 0x4" ::: "memory");
        } else {
            asm volatile("s_wait_asynccnt 0x0" ::: "memory");
        }
        __syncthreads();                              // buf[k&1] visible to all
        v16bf ah = load_a_frag(&sAh[buf][0], lane, wm * 16);
        v16bf al = load_a_frag(&sAl[buf][0], lane, wm * 16);
#pragma unroll
        for (int s = 0; s < 2; ++s) {
            v16bf bh = load_b_frag(&sBh[buf][0], lane, wn * 32 + s * 16);
            v16bf bl = load_b_frag(&sBl[buf][0], lane, wn * 32 + s * 16);
            acc[s] = wmma_bf16(ah, bh, acc[s]);   // hi*hi
            acc[s] = wmma_bf16(ah, bl, acc[s]);   // hi*lo
            acc[s] = wmma_bf16(al, bh, acc[s]);   // lo*hi
        }
        __syncthreads();   // all reads of buf done before it is re-filled
    }
}

// ---------------------------------------------------------------------------
// init: split U into hi/lo (both [P][K] and transposed [K][P]); zero R state.
// ---------------------------------------------------------------------------
__global__ void ista_init(const float* __restrict__ U,
                          float* R, unsigned short* Rhi, unsigned short* Rlo,
                          unsigned short* Uhi, unsigned short* Ulo,
                          unsigned short* Uthi, unsigned short* Utlo,
                          unsigned* flag, unsigned* cnt) {
    int gid = blockIdx.x * 256 + threadIdx.x;            // 0 .. 1048575
    R[gid]   = 0.0f;
    Rhi[gid] = 0;
    Rlo[gid] = 0;
    if (gid < NP * NK) {
        float u = U[gid];
        unsigned short hi = f2bf(u);
        unsigned short lo = f2bf(u - bf2f(hi));
        Uhi[gid] = hi;
        Ulo[gid] = lo;
        int p = gid >> 8;            // pixel row
        int j = gid & 255;           // code col
        Uthi[j * NP + p] = hi;
        Utlo[j * NP + p] = lo;
    }
    if (gid == 0) { *flag = 0u; *cnt = 0u; }
}

// ---------------------------------------------------------------------------
// predres: res = img - R@U^T as bf16 hi/lo.  grid (16, 64), block 256.
// ---------------------------------------------------------------------------
__global__ void ista_predres(const float* __restrict__ img,
                             const unsigned short* __restrict__ Rhi,
                             const unsigned short* __restrict__ Rlo,
                             const unsigned short* __restrict__ Uhi,
                             const unsigned short* __restrict__ Ulo,
                             unsigned short* __restrict__ reshi,
                             unsigned short* __restrict__ reslo,
                             const unsigned* __restrict__ flag) {
    if (*flag) return;
    const int tid = threadIdx.x, lane = tid & 31, w = tid >> 5;
    const int wm = w & 3, wn = w >> 2;
    const int bm = blockIdx.y, bn = blockIdx.x;

    v8f z = {0.f,0.f,0.f,0.f,0.f,0.f,0.f,0.f};
    v8f acc[2] = {z, z};
    gemm64x64(Rhi, Rlo, NK, bm * 64, Uhi, Ulo, NK, bn * 64, NK, acc);

    const int mb = (lane >> 4) * 8;
    const int n  = lane & 15;
#pragma unroll
    for (int s = 0; s < 2; ++s)
#pragma unroll
        for (int i = 0; i < 8; ++i) {
            int row = bm * 64 + wm * 16 + mb + i;
            int col = bn * 64 + wn * 32 + s * 16 + n;
            size_t idx = (size_t)row * NP + col;
            float r = img[idx] - acc[s][i];
            unsigned short hi = f2bf(r);
            reshi[idx] = hi;
            reslo[idx] = f2bf(r - bf2f(hi));
        }
}

// ---------------------------------------------------------------------------
// gradupd: G = res@U ; R = soft(R + 0.2*G) ; deterministic convergence test.
// grid (4, 64), block 256.  256 blocks total.
// ---------------------------------------------------------------------------
__global__ void ista_gradupd(const unsigned short* __restrict__ reshi,
                             const unsigned short* __restrict__ reslo,
                             const unsigned short* __restrict__ Uthi,
                             const unsigned short* __restrict__ Utlo,
                             float* __restrict__ R,
                             unsigned short* __restrict__ Rhi,
                             unsigned short* __restrict__ Rlo,
                             float* __restrict__ partials,
                             unsigned* flag, unsigned* cnt) {
    if (*flag) return;
    __shared__ float red0[256], red1[256];
    __shared__ int isLast;

    const int tid = threadIdx.x, lane = tid & 31, w = tid >> 5;
    const int wm = w & 3, wn = w >> 2;
    const int bm = blockIdx.y, bn = blockIdx.x;

    v8f z = {0.f,0.f,0.f,0.f,0.f,0.f,0.f,0.f};
    v8f acc[2] = {z, z};
    gemm64x64(reshi, reslo, NP, bm * 64, Uthi, Utlo, NP, bn * 64, NP, acc);

    // prox step + local norm accumulation
    const int mb = (lane >> 4) * 8;
    const int n  = lane & 15;
    float dsum = 0.f, ssum = 0.f;
#pragma unroll
    for (int s = 0; s < 2; ++s)
#pragma unroll
        for (int i = 0; i < 8; ++i) {
            int row = bm * 64 + wm * 16 + mb + i;
            int col = bn * 64 + wn * 32 + s * 16 + n;
            size_t idx = (size_t)row * NK + col;
            float rold = R[idx];
            float x  = rold + LR2 * acc[s][i];
            float ax = fabsf(x) - LMDA;
            float rn = (ax > 0.f) ? copysignf(ax, x) : 0.f;
            R[idx] = rn;
            unsigned short hi = f2bf(rn);
            Rhi[idx] = hi;
            Rlo[idx] = f2bf(rn - bf2f(hi));
            float dr = rn - rold;
            dsum += dr * dr;
            ssum += rold * rold;
        }

    // deterministic block tree reduction
    red0[tid] = dsum; red1[tid] = ssum;
    __syncthreads();
    for (int off = 128; off > 0; off >>= 1) {
        if (tid < off) { red0[tid] += red0[tid + off]; red1[tid] += red1[tid + off]; }
        __syncthreads();
    }
    const int blin = blockIdx.y * 4 + blockIdx.x;   // 0..255
    if (tid == 0) {
        partials[2 * blin]     = red0[0];
        partials[2 * blin + 1] = red1[0];
        __threadfence();
        unsigned t = atomicAdd(cnt, 1u);
        isLast = (t == 255u);
    }
    __syncthreads();
    if (isLast) {   // fixed-order final reduction -> bit-deterministic
        __threadfence();
        red0[tid] = partials[2 * tid];
        red1[tid] = partials[2 * tid + 1];
        __syncthreads();
        for (int off = 128; off > 0; off >>= 1) {
            if (tid < off) { red0[tid] += red0[tid + off]; red1[tid] += red1[tid + off]; }
            __syncthreads();
        }
        if (tid == 0) {
            float D = red0[0], S = red1[0];
            // ratio < TOL  <=>  D < TOL^2 * S ; S==0 or NaN -> false -> continue
            if (D < TOL2 * S) *flag = 1u;
            *cnt = 0u;
        }
    }
}

// ---------------------------------------------------------------------------
// output: d_out = R@U^T in f32.  grid (16, 64), block 256.
// ---------------------------------------------------------------------------
__global__ void ista_output(const unsigned short* __restrict__ Rhi,
                            const unsigned short* __restrict__ Rlo,
                            const unsigned short* __restrict__ Uhi,
                            const unsigned short* __restrict__ Ulo,
                            float* __restrict__ out) {
    const int tid = threadIdx.x, lane = tid & 31, w = tid >> 5;
    const int wm = w & 3, wn = w >> 2;
    const int bm = blockIdx.y, bn = blockIdx.x;

    v8f z = {0.f,0.f,0.f,0.f,0.f,0.f,0.f,0.f};
    v8f acc[2] = {z, z};
    gemm64x64(Rhi, Rlo, NK, bm * 64, Uhi, Ulo, NK, bn * 64, NK, acc);

    const int mb = (lane >> 4) * 8;
    const int n  = lane & 15;
#pragma unroll
    for (int s = 0; s < 2; ++s)
#pragma unroll
        for (int i = 0; i < 8; ++i) {
            int row = bm * 64 + wm * 16 + mb + i;
            int col = bn * 64 + wn * 32 + s * 16 + n;
            out[(size_t)row * NP + col] = acc[s][i];
        }
}

// ---------------------------------------------------------------------------
extern "C" void kernel_launch(void* const* d_in, const int* in_sizes, int n_in,
                              void* d_out, int out_size, void* d_ws, size_t ws_size,
                              hipStream_t stream) {
    (void)in_sizes; (void)n_in; (void)out_size; (void)ws_size;
    const float* img = (const float*)d_in[0];   // [4096, 1024]
    const float* U   = (const float*)d_in[1];   // [1024, 256]
    float* out = (float*)d_out;                 // [4096, 1024]

    char* ws = (char*)d_ws;
    size_t o = 0;
    float*          R     = (float*)(ws + o);          o += (size_t)NB * NK * 4;
    unsigned short* Rhi   = (unsigned short*)(ws + o); o += (size_t)NB * NK * 2;
    unsigned short* Rlo   = (unsigned short*)(ws + o); o += (size_t)NB * NK * 2;
    unsigned short* Uhi   = (unsigned short*)(ws + o); o += (size_t)NP * NK * 2;
    unsigned short* Ulo   = (unsigned short*)(ws + o); o += (size_t)NP * NK * 2;
    unsigned short* Uthi  = (unsigned short*)(ws + o); o += (size_t)NP * NK * 2;
    unsigned short* Utlo  = (unsigned short*)(ws + o); o += (size_t)NP * NK * 2;
    unsigned short* reshi = (unsigned short*)(ws + o); o += (size_t)NB * NP * 2;
    unsigned short* reslo = (unsigned short*)(ws + o); o += (size_t)NB * NP * 2;
    float*          part  = (float*)(ws + o);          o += 256 * 2 * 4;
    unsigned*       flag  = (unsigned*)(ws + o);       o += 4;
    unsigned*       cnt   = (unsigned*)(ws + o);       o += 4;

    ista_init<<<dim3((NB * NK) / 256), 256, 0, stream>>>(
        U, R, Rhi, Rlo, Uhi, Ulo, Uthi, Utlo, flag, cnt);

    for (int it = 0; it < MAXIT; ++it) {
        ista_predres<<<dim3(NP / 64, NB / 64), 256, 0, stream>>>(
            img, Rhi, Rlo, Uhi, Ulo, reshi, reslo, flag);
        ista_gradupd<<<dim3(NK / 64, NB / 64), 256, 0, stream>>>(
            reshi, reslo, Uthi, Utlo, R, Rhi, Rlo, part, flag, cnt);
    }

    ista_output<<<dim3(NP / 64, NB / 64), 256, 0, stream>>>(
        Rhi, Rlo, Uhi, Ulo, out);
}
